// Encoder_25297357374048
// MI455X (gfx1250) — matrix operand order repaired and checked by reading, not compiled, not run on hardware
//
#include <hip/hip_runtime.h>
#include <hip/hip_bf16.h>

// ---------------------------------------------------------------------------
// CDNA5 / gfx1250 transformer encoder.
//  - all matmuls through v_wmma_f32_16x16x32_f16 (f16 operands, f32 accum)
//  - weights pre-swizzled to WMMA-B fragment order (contiguous b128 loads)
//  - GEMM A-panels staged by the Tensor Data Mover (tensor_load_to_lds,
//    double-buffered LDS, s_wait_tensorcnt) per CDNA5 ISA
//  - fused attention keeps a full 16x4096 f32 score slab in 320KB WGP LDS
// ---------------------------------------------------------------------------

typedef _Float16 f16t;
typedef __attribute__((ext_vector_type(16))) _Float16 v16h;
typedef __attribute__((ext_vector_type(8)))  float    v8f;
typedef __attribute__((ext_vector_type(4)))  unsigned int u32x4;
typedef __attribute__((ext_vector_type(8)))  unsigned int u32x8;

#define NTOK   4096
#define EDIM   1024
#define NHEAD  16
#define HDIM   32
#define NLAYER 6
#define FDIM   2048
#define ATT_SCALE 0.17677669529663687f   // 1/sqrt(32)

__device__ __forceinline__ v8f wmma_f16(v16h a, v16h b, v8f c) {
  return __builtin_amdgcn_wmma_f32_16x16x32_f16(
      false, a, false, b, (short)0, c, false, false);
}

// A fragment (16x32 MxK, row-major, leading dim ld).
__device__ __forceinline__ v16h load_a_f16(const f16t* base, int ld) {
  int lane = threadIdx.x & 31;
  const f16t* p = base + (size_t)(lane & 15) * ld + (lane >> 4) * 16;
  v16h a;
#pragma unroll
  for (int j = 0; j < 8; ++j) a[j] = p[j];
#pragma unroll
  for (int j = 0; j < 8; ++j) a[8 + j] = p[8 + j];
  return a;
}

// B fragment from swizzled storage: tile = 32 lanes x 16 contiguous halfs.
__device__ __forceinline__ v16h load_b_swz(const f16t* tile) {
  const f16t* p = tile + (size_t)(threadIdx.x & 31) * 16;
  v16h b;
#pragma unroll
  for (int j = 0; j < 16; ++j) b[j] = p[j];
  return b;
}

// ---------------------------------------------------------------------------
// Convert f32 weight [K][Nc] -> f16 in WMMA-B fragment order.
__global__ __launch_bounds__(32) void swz_kernel(
    const float* __restrict__ src, f16t* __restrict__ dst,
    int Nc, size_t batch_stride) {
  int kt = blockIdx.x, nt = blockIdx.y, b = blockIdx.z;
  int lane = threadIdx.x;
  const float* s = src + (size_t)b * batch_stride;
  f16t* d = dst + (size_t)b * batch_stride +
            ((size_t)(kt * (Nc >> 4) + nt) * 32 + lane) * 16;
  int srow = kt * 32 + (lane >> 4) * 16;
  int scol = nt * 16 + (lane & 15);
#pragma unroll
  for (int j = 0; j < 16; ++j)
    d[j] = (f16t)s[(size_t)(srow + j) * Nc + scol];
}

// ---------------------------------------------------------------------------
// z0 = table[context] + pos ; also f16 mirror
__global__ __launch_bounds__(256) void embed_kernel(
    const int* __restrict__ ctx, const float* __restrict__ table,
    const float* __restrict__ pos, float* __restrict__ z, f16t* __restrict__ zh) {
  int n = blockIdx.x;
  int t = ctx[n];
  for (int i = threadIdx.x; i < EDIM; i += 256) {
    float v = table[(size_t)t * EDIM + i] + pos[(size_t)n * EDIM + i];
    z[(size_t)n * EDIM + i]  = v;
    zh[(size_t)n * EDIM + i] = (f16t)v;
  }
}

// ---------------------------------------------------------------------------
// Q/K/V projection, grid = (NTOK/16, NHEAD, 3), 1 wave/block.
__global__ __launch_bounds__(32) void qkv_kernel(
    const f16t* __restrict__ Z, const f16t* __restrict__ Wswz,
    f16t* __restrict__ QK, f16t* __restrict__ Vt) {
  int row0  = blockIdx.x * 16;
  int h     = blockIdx.y;
  int which = blockIdx.z;
  const f16t* W = Wswz + ((size_t)which * NHEAD + h) * (EDIM * HDIM);

  v8f acc0 = {}, acc1 = {};
  for (int k = 0; k < EDIM; k += 32) {
    v16h a = load_a_f16(Z + (size_t)row0 * EDIM + k, EDIM);
    int kt = k >> 5;
    const f16t* btile = W + (size_t)(kt * 2) * 512;
    v16h b0 = load_b_swz(btile);
    v16h b1 = load_b_swz(btile + 512);
    if (k + 32 < EDIM) __builtin_prefetch(btile + 1024, 0, 3);
    acc0 = wmma_f16(a, b0, acc0);
    acc1 = wmma_f16(a, b1, acc1);
  }
  int lane = threadIdx.x & 31;
  int c = lane & 15, g8 = (lane >> 4) * 8;
  if (which == 2) {
    f16t* OutT = Vt + (size_t)h * HDIM * NTOK;   // [32][NTOK]
#pragma unroll
    for (int g = 0; g < 8; ++g) {
      int m = row0 + g + g8;
      OutT[(size_t)c * NTOK + m]        = (f16t)acc0[g];
      OutT[(size_t)(16 + c) * NTOK + m] = (f16t)acc1[g];
    }
  } else {
    f16t* Out = QK + ((size_t)which * NHEAD + h) * ((size_t)NTOK * HDIM);
#pragma unroll
    for (int g = 0; g < 8; ++g) {
      int m = row0 + g + g8;
      Out[(size_t)m * HDIM + c]      = (f16t)acc0[g];
      Out[(size_t)m * HDIM + 16 + c] = (f16t)acc1[g];
    }
  }
}

// ---------------------------------------------------------------------------
// Fused attention: one workgroup (4 waves) per (head, 16-query-row block).
__global__ __launch_bounds__(128) void attn_kernel(
    const f16t* __restrict__ Q, const f16t* __restrict__ K,
    const f16t* __restrict__ Vt, f16t* __restrict__ O) {
  extern __shared__ float smem[];
  float* S    = smem;                    // 16*4096 scores -> probs in place
  float* red  = smem + 16 * NTOK;
  float* Ored = red + 160;

  int h    = blockIdx.y;
  int row0 = blockIdx.x * 16;
  int wave = threadIdx.x >> 5;
  int lane = threadIdx.x & 31;
  int c = lane & 15, grp = lane >> 4;

  const f16t* Qh = Q + (size_t)h * NTOK * HDIM;
  const f16t* Kh = K + ((size_t)NHEAD + h) * ((size_t)NTOK * HDIM);
  const f16t* Vh = Vt + (size_t)h * HDIM * NTOK;   // [32][NTOK]

  v16h aq = load_a_f16(Qh + (size_t)row0 * HDIM, HDIM);
  for (int kb = wave * 32; kb < NTOK; kb += 128) {
    v16h b0, b1;
#pragma unroll
    for (int j = 0; j < 16; ++j) {
      b0[j] = Kh[(size_t)(kb + c) * HDIM + grp * 16 + j];
      b1[j] = Kh[(size_t)(kb + 16 + c) * HDIM + grp * 16 + j];
    }
    v8f zz = {};
    v8f s0 = wmma_f16(aq, b0, zz);
    v8f s1 = wmma_f16(aq, b1, zz);
#pragma unroll
    for (int g = 0; g < 8; ++g) {
      int m = g + grp * 8;
      S[m * NTOK + kb + c]      = s0[g] * ATT_SCALE;
      S[m * NTOK + kb + 16 + c] = s1[g] * ATT_SCALE;
    }
  }
  __syncthreads();

  {
    int r = threadIdx.x >> 3, seg = threadIdx.x & 7;
    float* Srow = S + r * NTOK + seg * 512;
    float mx = -3.4e38f;
    for (int i = 0; i < 512; ++i) mx = fmaxf(mx, Srow[i]);
    red[r * 8 + seg] = mx;
    __syncthreads();
    if (seg == 0) {
      float m2 = red[r * 8];
      for (int i = 1; i < 8; ++i) m2 = fmaxf(m2, red[r * 8 + i]);
      red[128 + r] = m2;
    }
    __syncthreads();
    float rowmax = red[128 + r];
    float sum = 0.f;
    for (int i = 0; i < 512; ++i) { float e = expf(Srow[i] - rowmax); Srow[i] = e; sum += e; }
    __syncthreads();
    red[r * 8 + seg] = sum;
    __syncthreads();
    if (seg == 0) {
      float s2 = 0.f;
      for (int i = 0; i < 8; ++i) s2 += red[r * 8 + i];
      red[144 + r] = s2;
    }
    __syncthreads();
    float inv = 1.0f / red[144 + r];
    for (int i = 0; i < 512; ++i) Srow[i] *= inv;
  }
  __syncthreads();

  v8f o0 = {}, o1 = {};
  for (int kb = wave * 32; kb < NTOK; kb += 128) {
    v16h ap;
    {
      const float* p = S + (size_t)(lane & 15) * NTOK + kb + grp * 16;
#pragma unroll
      for (int j = 0; j < 8; ++j) ap[j] = (f16t)p[j];
#pragma unroll
      for (int j = 0; j < 8; ++j) ap[8 + j] = (f16t)p[8 + j];
    }
    v16h b0, b1;
#pragma unroll
    for (int j = 0; j < 16; ++j) {
      b0[j] = Vh[(size_t)c * NTOK + kb + grp * 16 + j];
      b1[j] = Vh[(size_t)(16 + c) * NTOK + kb + grp * 16 + j];
    }
    o0 = wmma_f16(ap, b0, o0);
    o1 = wmma_f16(ap, b1, o1);
  }
#pragma unroll
  for (int g = 0; g < 8; ++g) {
    int m = g + grp * 8;
    Ored[wave * 512 + m * 32 + c]      = o0[g];
    Ored[wave * 512 + m * 32 + 16 + c] = o1[g];
  }
  __syncthreads();
  for (int idx = threadIdx.x; idx < 512; idx += 128) {
    float v = Ored[idx] + Ored[512 + idx] + Ored[1024 + idx] + Ored[1536 + idx];
    int m = idx >> 5, vv = idx & 31;
    O[(size_t)(row0 + m) * (NHEAD * HDIM) + h * HDIM + vv] = (f16t)v;
  }
}

// ---------------------------------------------------------------------------
// WMMA GEMM: C[M x Nc] = act(A_f16[M x K] @ B + bias), B in swizzled order.
// Block = 4 waves, macro tile 64 x 256; each wave: 4x4 grid of 16x16 tiles
// (16 WMMA per K-step).  The shared 64x32 A panel is staged by the Tensor
// Data Mover: one tensor_load_to_lds per K-step, double-buffered, TENSORcnt.
__global__ __launch_bounds__(128) void wgemm_kernel(
    const f16t* __restrict__ A, const f16t* __restrict__ Bswz,
    const float* __restrict__ bias, float* __restrict__ Cf32,
    f16t* __restrict__ Cf16, int M, int K, int Nc, int act) {
  __shared__ f16t Atile[2][64 * 32];
  int row0 = blockIdx.x * 64;
  int col0 = blockIdx.y * 256;
  int tid  = threadIdx.x;
  int wave = tid >> 5, lane = tid & 31;
  int ntiles = Nc >> 4;
  int nt0 = (col0 >> 4) + wave * 4;
  (void)M;

  v8f zero = {};
  v8f acc[4][4];
#pragma unroll
  for (int i = 0; i < 4; ++i)
#pragma unroll
    for (int j = 0; j < 4; ++j) acc[i][j] = zero;

  // TDM descriptor group 1 (constant): 2-byte elems, tile 32 x 64, row
  // stride = K elements; tensor dims huge so no OOB clamping applies.
  u32x8 g1;
  g1[0] = 0x00010000u;                   // data_size=2B; no multicast/pad/iter
  g1[1] = 0xFFFF0000u;                   // tensor_dim0 lo16 (0x7FFFFFFF)
  g1[2] = 0x7FFFu | (0xFFFFu << 16);     // tensor_dim0 hi16 | tensor_dim1 lo16
  g1[3] = 0x7FFFu | (32u << 16);         // tensor_dim1 hi16 | tile_dim0 = 32
  g1[4] = 64u;                           // tile_dim1 = 64, tile_dim2 = 0
  g1[5] = (unsigned)K;                   // tensor_dim0_stride lo32 (elements)
  g1[6] = 0u;
  g1[7] = 0u;

  auto tdm_issue = [&](int k, int buf) {
    unsigned long long ga =
        (unsigned long long)(size_t)(A + (size_t)row0 * K + k);
    unsigned lds = (unsigned)(size_t)(&Atile[buf][0]);
    u32x4 g0;
    g0[0] = 1u;                                           // count=1, user D#
    g0[1] = lds;                                          // lds_addr (bytes)
    g0[2] = (unsigned)ga;                                 // global_addr lo32
    g0[3] = ((unsigned)(ga >> 32) & 0x01FFFFFFu) | (2u << 30); // hi25 | type=2
    asm volatile("tensor_load_to_lds %0, %1" :: "s"(g0), "s"(g1) : "memory");
  };

  if (wave == 0) tdm_issue(0, 0);

  int nsteps = K >> 5;
  for (int s = 0; s < nsteps; ++s) {
    int cur = s & 1;
    if (wave == 0) __builtin_amdgcn_s_wait_tensorcnt(0);
    __syncthreads();                           // Atile[cur] ready everywhere
    if (wave == 0 && s + 1 < nsteps) tdm_issue((s + 1) << 5, cur ^ 1);

    v16h afr[4];
#pragma unroll
    for (int rt = 0; rt < 4; ++rt) {
      const f16t* p = &Atile[cur][(rt * 16 + (lane & 15)) * 32 + (lane >> 4) * 16];
#pragma unroll
      for (int j = 0; j < 8; ++j) afr[rt][j] = p[j];
#pragma unroll
      for (int j = 0; j < 8; ++j) afr[rt][8 + j] = p[8 + j];
    }
    const f16t* btile = Bswz + ((size_t)s * ntiles + nt0) * 512;
    if (s + 1 < nsteps) __builtin_prefetch(btile + (size_t)ntiles * 512, 0, 3);
#pragma unroll
    for (int ct = 0; ct < 4; ++ct) {
      v16h b = load_b_swz(btile + (size_t)ct * 512);
#pragma unroll
      for (int rt = 0; rt < 4; ++rt)
        acc[rt][ct] = wmma_f16(afr[rt], b, acc[rt][ct]);
    }
    // no trailing barrier: next iteration's barrier precedes the refill of
    // Atile[cur], so all reads above are safely ordered before overwrite.
  }

  int c = lane & 15, g8 = (lane >> 4) * 8;
#pragma unroll
  for (int ct = 0; ct < 4; ++ct) {
    int n = col0 + wave * 64 + ct * 16 + c;
    float bb = bias ? bias[n] : 0.f;
#pragma unroll
    for (int rt = 0; rt < 4; ++rt) {
#pragma unroll
      for (int g = 0; g < 8; ++g) {
        int m = row0 + rt * 16 + g + g8;
        float v = acc[rt][ct][g] + bb;
        if (act) v = (v > 0.f) ? v : 0.01f * v;
        if (Cf32) Cf32[(size_t)m * Nc + n] = v;
        if (Cf16) Cf16[(size_t)m * Nc + n] = (f16t)v;
      }
    }
  }
}

// ---------------------------------------------------------------------------
// out = layernorm(x + y) over E (unbiased std, no eps); f16 mirror optional.
__global__ __launch_bounds__(256) void add_ln_kernel(
    const float* __restrict__ x, const float* __restrict__ y,
    float* __restrict__ out, f16t* __restrict__ outh) {
  __shared__ float rbuf[256];
  int row = blockIdx.x, tid = threadIdx.x;
  const float* xr = x + (size_t)row * EDIM;
  const float* yr = y + (size_t)row * EDIM;
  float vals[4];
  float s = 0.f;
#pragma unroll
  for (int i = 0; i < 4; ++i) {
    vals[i] = xr[tid + i * 256] + yr[tid + i * 256];
    s += vals[i];
  }
  rbuf[tid] = s;
  __syncthreads();
  for (int st = 128; st > 0; st >>= 1) {
    if (tid < st) rbuf[tid] += rbuf[tid + st];
    __syncthreads();
  }
  float mean = rbuf[0] / (float)EDIM;
  __syncthreads();
  float s2 = 0.f;
#pragma unroll
  for (int i = 0; i < 4; ++i) { float d = vals[i] - mean; s2 += d * d; }
  rbuf[tid] = s2;
  __syncthreads();
  for (int st = 128; st > 0; st >>= 1) {
    if (tid < st) rbuf[tid] += rbuf[tid + st];
    __syncthreads();
  }
  float rstd = rsqrtf(rbuf[0] / (float)(EDIM - 1));
#pragma unroll
  for (int i = 0; i < 4; ++i) {
    float v = (vals[i] - mean) * rstd;
    out[(size_t)row * EDIM + tid + i * 256] = v;
    if (outh) outh[(size_t)row * EDIM + tid + i * 256] = (f16t)v;
  }
}

// ---------------------------------------------------------------------------
extern "C" void kernel_launch(void* const* d_in, const int* in_sizes, int n_in,
                              void* d_out, int out_size, void* d_ws, size_t ws_size,
                              hipStream_t stream) {
  (void)in_sizes; (void)n_in; (void)out_size; (void)ws_size;

  const int*   ctx   = (const int*)  d_in[0];
  const float* table = (const float*)d_in[1];
  const float* pos   = (const float*)d_in[2];
  const float* Wq    = (const float*)d_in[3];
  const float* Wk    = (const float*)d_in[4];
  const float* Wv    = (const float*)d_in[5];
  const float* Wo    = (const float*)d_in[6];
  const float* W1    = (const float*)d_in[7];
  const float* b1    = (const float*)d_in[8];
  const float* W2    = (const float*)d_in[9];
  const float* b2    = (const float*)d_in[10];
  float* out = (float*)d_out;

  char* ws = (char*)d_ws;
  size_t off = 0;
  auto carve = [&](size_t bytes) -> void* {
    void* p = ws + off;
    off += (bytes + 255) & ~(size_t)255;
    return p;
  };
  const size_t NE  = (size_t)NTOK * EDIM;
  const size_t HEV = (size_t)NHEAD * EDIM * HDIM;
  const size_t HNV = (size_t)NHEAD * NTOK * HDIM;

  float* z    = (float*)carve(NE * 4);
  float* an   = (float*)carve(NE * 4);
  float* tbuf = (float*)carve(NE * 4);
  f16t*  zh   = (f16t*) carve(NE * 2);
  f16t*  fbuf = (f16t*) carve((size_t)NTOK * FDIM * 2);
  f16t*  qkvW = (f16t*) carve(3 * HEV * 2);
  f16t*  wo16 = (f16t*) carve((size_t)NHEAD * HDIM * EDIM * 2);
  f16t*  w116 = (f16t*) carve((size_t)EDIM * FDIM * 2);
  f16t*  w216 = (f16t*) carve((size_t)FDIM * EDIM * 2);
  f16t*  QK   = (f16t*) carve(2 * HNV * 2);
  f16t*  Vt   = (f16t*) carve(HNV * 2);
  f16t*  Obuf = (f16t*) carve((size_t)NTOK * NHEAD * HDIM * 2);

  const int attn_lds = (16 * NTOK + 160 + 4 * 16 * 32) * (int)sizeof(float);
  (void)hipFuncSetAttribute(reinterpret_cast<const void*>(&attn_kernel),
                            hipFuncAttributeMaxDynamicSharedMemorySize, attn_lds);

  embed_kernel<<<NTOK, 256, 0, stream>>>(ctx, table, pos, z, zh);

  for (int l = 0; l < NLAYER; ++l) {
    const float* Wq_l = Wq + (size_t)l * HEV;
    const float* Wk_l = Wk + (size_t)l * HEV;
    const float* Wv_l = Wv + (size_t)l * HEV;
    const float* Wo_l = Wo + (size_t)l * NHEAD * HDIM * EDIM;
    const float* W1_l = W1 + (size_t)l * EDIM * FDIM;
    const float* b1_l = b1 + (size_t)l * FDIM;
    const float* W2_l = W2 + (size_t)l * FDIM * EDIM;
    const float* b2_l = b2 + (size_t)l * EDIM;

    swz_kernel<<<dim3(EDIM / 32, HDIM / 16, NHEAD), 32, 0, stream>>>(
        Wq_l, qkvW,           HDIM, (size_t)EDIM * HDIM);
    swz_kernel<<<dim3(EDIM / 32, HDIM / 16, NHEAD), 32, 0, stream>>>(
        Wk_l, qkvW + HEV,     HDIM, (size_t)EDIM * HDIM);
    swz_kernel<<<dim3(EDIM / 32, HDIM / 16, NHEAD), 32, 0, stream>>>(
        Wv_l, qkvW + 2 * HEV, HDIM, (size_t)EDIM * HDIM);
    swz_kernel<<<dim3((NHEAD * HDIM) / 32, EDIM / 16, 1), 32, 0, stream>>>(
        Wo_l, wo16, EDIM, 0);
    swz_kernel<<<dim3(EDIM / 32, FDIM / 16, 1), 32, 0, stream>>>(
        W1_l, w116, FDIM, 0);
    swz_kernel<<<dim3(FDIM / 32, EDIM / 16, 1), 32, 0, stream>>>(
        W2_l, w216, EDIM, 0);

    qkv_kernel<<<dim3(NTOK / 16, NHEAD, 3), 32, 0, stream>>>(zh, qkvW, QK, Vt);

    attn_kernel<<<dim3(NTOK / 16, NHEAD), 128, attn_lds, stream>>>(QK, QK, Vt, Obuf);

    wgemm_kernel<<<dim3(NTOK / 64, EDIM / 256), 128, 0, stream>>>(
        Obuf, wo16, nullptr, tbuf, nullptr, NTOK, NHEAD * HDIM, EDIM, 0);

    add_ln_kernel<<<NTOK, 256, 0, stream>>>(z, tbuf, an, zh);

    wgemm_kernel<<<dim3(NTOK / 64, FDIM / 256), 128, 0, stream>>>(
        zh, w116, b1_l, nullptr, fbuf, NTOK, EDIM, FDIM, 1);

    wgemm_kernel<<<dim3(NTOK / 64, EDIM / 256), 128, 0, stream>>>(
        fbuf, w216, b2_l, tbuf, nullptr, NTOK, FDIM, EDIM, 0);

    float* zout = (l == NLAYER - 1) ? out : z;
    add_ln_kernel<<<NTOK, 256, 0, stream>>>(an, tbuf, zout, zh);
  }
}
